// GraphNetBlock_31533649887539
// MI455X (gfx1250) — matrix operand order repaired and checked by reading, not compiled, not run on hardware
//
#include <hip/hip_runtime.h>
#include <hip/hip_bf16.h>

typedef __attribute__((ext_vector_type(8)))  float  f32x8;
typedef __attribute__((ext_vector_type(4)))  float  f32x4;
typedef __attribute__((ext_vector_type(16))) __bf16 bf16x16;
typedef __attribute__((ext_vector_type(8)))  __bf16 bf16x8;

#define HDIM 128
#define WAVES_PER_BLOCK 4
#define MT 2                      // 16-row M-tiles per wave (32 rows/wave)
#define ROWS_PER_WAVE (16 * MT)

// ---------------------------------------------------------------------------
// helpers
// ---------------------------------------------------------------------------
static __device__ __forceinline__ void zero_acc(f32x8 (&acc)[MT][8]) {
#pragma unroll
  for (int t = 0; t < MT; ++t)
#pragma unroll
    for (int nt = 0; nt < 8; ++nt)
#pragma unroll
      for (int v = 0; v < 8; ++v) acc[t][nt][v] = 0.0f;
}

// One k-tile (K=32): load each pre-swizzled B fragment once, feed MT A tiles.
static __device__ __forceinline__ void wmma_tiles(const bf16x16 (&a)[MT],
                                                  const bf16x16* __restrict__ wfrag,
                                                  int kt, int lane, f32x8 (&acc)[MT][8]) {
#pragma unroll
  for (int nt = 0; nt < 8; ++nt) {
    bf16x16 b = wfrag[(size_t)(kt * 8 + nt) * 32 + lane];
#pragma unroll
    for (int t = 0; t < MT; ++t) {
      acc[t][nt] = __builtin_amdgcn_wmma_f32_16x16x32_bf16(
          false, a[t], false, b, (short)0, acc[t][nt], false, false);
    }
  }
}

// Hidden layer (K=128): A fragments from row-major bf16 LDS tile (MT*16 x 128).
static __device__ __forceinline__ void layer_from_lds(const __bf16* __restrict__ hrow,
                                                      const bf16x16* __restrict__ wfrag,
                                                      int lane, f32x8 (&acc)[MT][8]) {
  const int m = lane & 15, hi = lane >> 4;
#pragma unroll
  for (int kt = 0; kt < 4; ++kt) {
    bf16x16 a[MT];
#pragma unroll
    for (int t = 0; t < MT; ++t) {
      const __bf16* p = hrow + (t * 16 + m) * HDIM + kt * 32 + hi * 8;
      bf16x8 a0 = *(const bf16x8*)(p);        // K = kbase .. kbase+7
      bf16x8 a1 = *(const bf16x8*)(p + 16);   // K = kbase+16 .. kbase+23
#pragma unroll
      for (int i = 0; i < 8; ++i) { a[t][i] = a0[i]; a[t][i + 8] = a1[i]; }
    }
    wmma_tiles(a, wfrag, kt, lane, acc);
  }
}

// Bias + ReLU in C layout, store bf16 activation tiles to LDS (row-major).
static __device__ __forceinline__ void bias_relu_store(f32x8 (&acc)[MT][8],
                                                       const float* __restrict__ b,
                                                       __bf16* __restrict__ hrow, int lane) {
  const int m = lane & 15, hi = lane >> 4;
#pragma unroll
  for (int nt = 0; nt < 8; ++nt) {
    const float bb = b[nt * 16 + m];
#pragma unroll
    for (int t = 0; t < MT; ++t)
#pragma unroll
      for (int v = 0; v < 8; ++v) {
        float x = acc[t][nt][v] + bb;
        x = x > 0.0f ? x : 0.0f;
        hrow[(t * 16 + v + 8 * hi) * HDIM + nt * 16 + m] = (__bf16)x;
      }
  }
}

// Row-wise LayerNorm stats. Row (t, v+8*hi) lives across 16 lanes (same hi).
static __device__ __forceinline__ void layernorm_stats(const f32x8 (&acc)[MT][8],
                                                       float (&mu)[MT][8], float (&rs)[MT][8]) {
#pragma unroll
  for (int t = 0; t < MT; ++t)
#pragma unroll
    for (int v = 0; v < 8; ++v) {
      float sm = 0.0f, sq = 0.0f;
#pragma unroll
      for (int nt = 0; nt < 8; ++nt) { float x = acc[t][nt][v]; sm += x; sq += x * x; }
#pragma unroll
      for (int off = 1; off < 16; off <<= 1) {
        sm += __shfl_xor(sm, off);
        sq += __shfl_xor(sq, off);
      }
      const float mean = sm * (1.0f / 128.0f);
      const float var  = sq * (1.0f / 128.0f) - mean * mean;
      mu[t][v] = mean;
      rs[t][v] = rsqrtf(var + 1e-5f);
    }
}

// ---------------------------------------------------------------------------
// prep kernels
// ---------------------------------------------------------------------------
// Convert fp32 weight (K x 128, row-major) into bf16 WMMA-B fragments.
// dst element idx -> fragment (kt,nt), lane l, element e:
//   B[kt*32 + (e>>3)*16 + (l>>4)*8 + (e&7)][nt*16 + (l&15)]
__global__ void swizzle_weights(const float* __restrict__ src, __bf16* __restrict__ dst,
                                int total) {
  int idx = blockIdx.x * 256 + threadIdx.x;
  if (idx >= total) return;
  int frag = idx >> 9;        // 512 bf16 per fragment
  int w    = idx & 511;
  int l    = w >> 4;
  int e    = w & 15;
  int kt   = frag >> 3;
  int nt   = frag & 7;
  int K    = kt * 32 + (e >> 3) * 16 + (l >> 4) * 8 + (e & 7);
  int N    = nt * 16 + (l & 15);
  dst[idx] = (__bf16)src[K * HDIM + N];
}

__global__ void zero_f32(float* __restrict__ p, int n) {
  int i = (blockIdx.x * 256 + threadIdx.x) * 4;
  if (i < n) {
    f32x4 z; z[0] = 0.0f; z[1] = 0.0f; z[2] = 0.0f; z[3] = 0.0f;
    *(f32x4*)(p + i) = z;
  }
}

// ---------------------------------------------------------------------------
// edge update kernel: one wave == 32 edges, fused 384->128->128->128 + LN
// ---------------------------------------------------------------------------
__global__ __launch_bounds__(32 * WAVES_PER_BLOCK) void gnb_edge_kernel(
    const int* __restrict__ eidx, const float* __restrict__ nf, const float* __restrict__ ef,
    const __bf16* __restrict__ w0, const __bf16* __restrict__ w1, const __bf16* __restrict__ w2,
    const float* __restrict__ b0, const float* __restrict__ b1, const float* __restrict__ b2,
    const float* __restrict__ g, const float* __restrict__ be,
    float* __restrict__ oedge, float* __restrict__ agg, int E) {
  __shared__ __bf16 hbuf[WAVES_PER_BLOCK * ROWS_PER_WAVE * HDIM];   // 32 KB
  const int lane = threadIdx.x & 31;
  const int wave = threadIdx.x >> 5;
  const int numTiles = (E + ROWS_PER_WAVE - 1) / ROWS_PER_WAVE;
  const int tile = blockIdx.x * WAVES_PER_BLOCK + wave;
  if (tile >= numTiles) return;           // wave-uniform: EXEC stays all-1s
  const int e0 = tile * ROWS_PER_WAVE;
  const int m = lane & 15, hi = lane >> 4;
  __bf16* hrow = hbuf + wave * (ROWS_PER_WAVE * HDIM);

  const float* segp[MT][3];
#pragma unroll
  for (int t = 0; t < MT; ++t) {
    int em = e0 + t * 16 + m;
    em = em < E ? em : E - 1;             // clamp partial tile (gather only)
    const int s = eidx[em];
    const int r = eidx[E + em];
    segp[t][0] = nf + (size_t)s * HDIM;
    segp[t][1] = nf + (size_t)r * HDIM;
    segp[t][2] = ef + (size_t)em * HDIM;
  }

  f32x8 acc[MT][8];
  zero_acc(acc);

  // layer 0: K = 384 over [sender | receiver | edge], gather A directly
#pragma unroll
  for (int kt = 0; kt < 12; ++kt) {
    const int c = ((kt & 3) << 5) + hi * 8;
    bf16x16 a[MT];
#pragma unroll
    for (int t = 0; t < MT; ++t) {
      const float* base = segp[t][kt >> 2];
      f32x8 f0 = *(const f32x8*)(base + c);
      f32x8 f1 = *(const f32x8*)(base + c + 16);
#pragma unroll
      for (int i = 0; i < 8; ++i) { a[t][i] = (__bf16)f0[i]; a[t][i + 8] = (__bf16)f1[i]; }
    }
    wmma_tiles(a, (const bf16x16*)w0, kt, lane, acc);
  }
  bias_relu_store(acc, b0, hrow, lane);

  zero_acc(acc);
  layer_from_lds(hrow, (const bf16x16*)w1, lane, acc);
  bias_relu_store(acc, b1, hrow, lane);

  zero_acc(acc);
  layer_from_lds(hrow, (const bf16x16*)w2, lane, acc);
#pragma unroll
  for (int nt = 0; nt < 8; ++nt) {
    const float bb = b2[nt * 16 + m];
#pragma unroll
    for (int t = 0; t < MT; ++t)
#pragma unroll
      for (int v = 0; v < 8; ++v) acc[t][nt][v] += bb;
  }

  float mu[MT][8], rs[MT][8];
  layernorm_stats(acc, mu, rs);

#pragma unroll
  for (int nt = 0; nt < 8; ++nt) {
    const int col = nt * 16 + m;
    const float gv = g[col], bv = be[col];
#pragma unroll
    for (int t = 0; t < MT; ++t)
#pragma unroll
      for (int v = 0; v < 8; ++v) {
        const int e = e0 + t * 16 + v + 8 * hi;
        if (e < E) {
          const float y = (acc[t][nt][v] - mu[t][v]) * rs[t][v] * gv + bv;
          oedge[(size_t)e * HDIM + col] = y + ef[(size_t)e * HDIM + col];
          const int rr = eidx[E + e];
          atomicAdd(agg + (size_t)rr * HDIM + col, y);   // segment_sum
        }
      }
  }
}

// ---------------------------------------------------------------------------
// node update kernel: one wave == 32 nodes, fused 256->128->128->128 + LN
// ---------------------------------------------------------------------------
__global__ __launch_bounds__(32 * WAVES_PER_BLOCK) void gnb_node_kernel(
    const float* __restrict__ nf, const float* __restrict__ agg,
    const __bf16* __restrict__ w0, const __bf16* __restrict__ w1, const __bf16* __restrict__ w2,
    const float* __restrict__ b0, const float* __restrict__ b1, const float* __restrict__ b2,
    const float* __restrict__ g, const float* __restrict__ be,
    float* __restrict__ onode, int N) {
  __shared__ __bf16 hbuf[WAVES_PER_BLOCK * ROWS_PER_WAVE * HDIM];   // 32 KB
  const int lane = threadIdx.x & 31;
  const int wave = threadIdx.x >> 5;
  const int numTiles = (N + ROWS_PER_WAVE - 1) / ROWS_PER_WAVE;
  const int tile = blockIdx.x * WAVES_PER_BLOCK + wave;
  if (tile >= numTiles) return;
  const int n0 = tile * ROWS_PER_WAVE;
  const int m = lane & 15, hi = lane >> 4;
  __bf16* hrow = hbuf + wave * (ROWS_PER_WAVE * HDIM);

  const float* segp[MT][2];
#pragma unroll
  for (int t = 0; t < MT; ++t) {
    int nm = n0 + t * 16 + m;
    nm = nm < N ? nm : N - 1;
    segp[t][0] = nf  + (size_t)nm * HDIM;
    segp[t][1] = agg + (size_t)nm * HDIM;
  }

  f32x8 acc[MT][8];
  zero_acc(acc);

  // layer 0: K = 256 over [node | aggregated edges]
#pragma unroll
  for (int kt = 0; kt < 8; ++kt) {
    const int c = ((kt & 3) << 5) + hi * 8;
    bf16x16 a[MT];
#pragma unroll
    for (int t = 0; t < MT; ++t) {
      const float* base = segp[t][kt >> 2];
      f32x8 f0 = *(const f32x8*)(base + c);
      f32x8 f1 = *(const f32x8*)(base + c + 16);
#pragma unroll
      for (int i = 0; i < 8; ++i) { a[t][i] = (__bf16)f0[i]; a[t][i + 8] = (__bf16)f1[i]; }
    }
    wmma_tiles(a, (const bf16x16*)w0, kt, lane, acc);
  }
  bias_relu_store(acc, b0, hrow, lane);

  zero_acc(acc);
  layer_from_lds(hrow, (const bf16x16*)w1, lane, acc);
  bias_relu_store(acc, b1, hrow, lane);

  zero_acc(acc);
  layer_from_lds(hrow, (const bf16x16*)w2, lane, acc);
#pragma unroll
  for (int nt = 0; nt < 8; ++nt) {
    const float bb = b2[nt * 16 + m];
#pragma unroll
    for (int t = 0; t < MT; ++t)
#pragma unroll
      for (int v = 0; v < 8; ++v) acc[t][nt][v] += bb;
  }

  float mu[MT][8], rs[MT][8];
  layernorm_stats(acc, mu, rs);

#pragma unroll
  for (int nt = 0; nt < 8; ++nt) {
    const int col = nt * 16 + m;
    const float gv = g[col], bv = be[col];
#pragma unroll
    for (int t = 0; t < MT; ++t)
#pragma unroll
      for (int v = 0; v < 8; ++v) {
        const int n = n0 + t * 16 + v + 8 * hi;
        if (n < N) {
          const float y = (acc[t][nt][v] - mu[t][v]) * rs[t][v] * gv + bv;
          onode[(size_t)n * HDIM + col] = y + nf[(size_t)n * HDIM + col];
        }
      }
  }
}

// ---------------------------------------------------------------------------
// launch
// ---------------------------------------------------------------------------
extern "C" void kernel_launch(void* const* d_in, const int* in_sizes, int n_in,
                              void* d_out, int out_size, void* d_ws, size_t ws_size,
                              hipStream_t stream) {
  (void)n_in; (void)out_size; (void)ws_size;
  const int*   eidx  = (const int*)d_in[0];
  const float* nf    = (const float*)d_in[1];
  const float* ef    = (const float*)d_in[2];
  const float* ew0   = (const float*)d_in[3];  const float* eb0   = (const float*)d_in[4];
  const float* ew1   = (const float*)d_in[5];  const float* eb1   = (const float*)d_in[6];
  const float* ew2   = (const float*)d_in[7];  const float* eb2   = (const float*)d_in[8];
  const float* eg    = (const float*)d_in[9];  const float* ebeta = (const float*)d_in[10];
  const float* nw0   = (const float*)d_in[11]; const float* nb0   = (const float*)d_in[12];
  const float* nw1   = (const float*)d_in[13]; const float* nb1   = (const float*)d_in[14];
  const float* nw2   = (const float*)d_in[15]; const float* nb2   = (const float*)d_in[16];
  const float* ng    = (const float*)d_in[17]; const float* nbeta = (const float*)d_in[18];

  const int E = in_sizes[0] / 2;
  const int N = in_sizes[1] / HDIM;

  // workspace: [ agg (N*128 f32) | swizzled bf16 weights ]
  float* agg = (float*)d_ws;
  size_t aggBytes = ((size_t)N * HDIM * sizeof(float) + 255) & ~(size_t)255;
  __bf16* wbase = (__bf16*)((char*)d_ws + aggBytes);
  __bf16* sew0 = wbase;
  __bf16* sew1 = sew0 + 384 * HDIM;
  __bf16* sew2 = sew1 + 128 * HDIM;
  __bf16* snw0 = sew2 + 128 * HDIM;
  __bf16* snw1 = snw0 + 256 * HDIM;
  __bf16* snw2 = snw1 + 128 * HDIM;

  auto sw = [&](const float* src, __bf16* dst, int K) {
    int total = K * HDIM;
    swizzle_weights<<<(total + 255) / 256, 256, 0, stream>>>(src, dst, total);
  };
  sw(ew0, sew0, 384); sw(ew1, sew1, 128); sw(ew2, sew2, 128);
  sw(nw0, snw0, 256); sw(nw1, snw1, 128); sw(nw2, snw2, 128);

  const int aggN = N * HDIM;
  zero_f32<<<(aggN / 4 + 255) / 256, 256, 0, stream>>>(agg, aggN);

  float* onode = (float*)d_out;
  float* oedge = onode + (size_t)N * HDIM;

  const int eTiles = (E + ROWS_PER_WAVE - 1) / ROWS_PER_WAVE;
  gnb_edge_kernel<<<(eTiles + WAVES_PER_BLOCK - 1) / WAVES_PER_BLOCK,
                    32 * WAVES_PER_BLOCK, 0, stream>>>(
      eidx, nf, ef, sew0, sew1, sew2, eb0, eb1, eb2, eg, ebeta, oedge, agg, E);

  const int nTiles = (N + ROWS_PER_WAVE - 1) / ROWS_PER_WAVE;
  gnb_node_kernel<<<(nTiles + WAVES_PER_BLOCK - 1) / WAVES_PER_BLOCK,
                    32 * WAVES_PER_BLOCK, 0, stream>>>(
      nf, agg, snw0, snw1, snw2, nb0, nb1, nb2, ng, nbeta, onode, N);
}
